// MultiPropMLP_24223615549690
// MI455X (gfx1250) — matrix operand order repaired and checked by reading, not compiled, not run on hardware
//
#include <hip/hip_runtime.h>
#include <hip/hip_bf16.h>

typedef __attribute__((ext_vector_type(16))) _Float16 v16h;
typedef __attribute__((ext_vector_type(8)))  _Float16 v8h;
typedef __attribute__((ext_vector_type(8)))  float    v8f;

#define N_SAMP  (8192*128)
#define D_IN    16
#define WIDTH   64
#define NK      8
#define NPAD    (N_SAMP + 128)      // each of 8 buckets padded to multiple of 16
#define MAXTILE (NPAD/16)           // 65544 tiles of 16 samples

__device__ __forceinline__ v8f wmma_f16(v16h a, v16h b, v8f c) {
  // D = A(16x32 f16) * B(32x16 f16) + C(16x16 f32)
  return __builtin_amdgcn_wmma_f32_16x16x32_f16(false, a, false, b, (short)0, c,
                                                false, false);
}

// ---------------------------------------------------------------------------
// prep: zero meta, fill perm with -1, convert + pre-shuffle weights into the
// exact per-lane WMMA B-fragment layout (one contiguous v16h per lane).
//   B 32x16 f16 layout assumed: N = lane&15, K = (lane<16 ? 0 : 8) + (elem&7) + (elem>=8 ? 16 : 0).
// ---------------------------------------------------------------------------
__global__ void prep_kernel(const float* __restrict__ W0, const float* __restrict__ W1,
                            int* __restrict__ meta, int* __restrict__ perm,
                            _Float16* __restrict__ W0p, _Float16* __restrict__ W1p) {
  int i = blockIdx.x * blockDim.x + threadIdx.x;
  if (i < 32) meta[i] = 0;
  if (i < NPAD) perm[i] = -1;
  if (i < NK*4*32*16) {            // W0 packed: [k][nt][lane][e], K padded 16->32
    int e = i & 15, lane = (i >> 4) & 31, nt = (i >> 9) & 3, k = i >> 11;
    int Kidx = ((lane & 16) ? 8 : 0) + (e & 7) + ((e & 8) ? 16 : 0);
    int n = nt*16 + (lane & 15);
    float v = (Kidx < D_IN) ? W0[(k*D_IN + Kidx)*WIDTH + n] : 0.0f;
    W0p[i] = (_Float16)v;
  }
  if (i < NK*2*4*32*16) {          // W1 packed: [k][kt][nt][lane][e]
    int e = i & 15, lane = (i >> 4) & 31, nt = (i >> 9) & 3, kt = (i >> 11) & 1, k = i >> 12;
    int Kidx = kt*32 + ((lane & 16) ? 8 : 0) + (e & 7) + ((e & 8) ? 16 : 0);
    int n = nt*16 + (lane & 15);
    W1p[i] = (_Float16)W1[(k*WIDTH + Kidx)*WIDTH + n];
  }
}

__global__ void count_kernel(const int* __restrict__ idxs, int* __restrict__ meta) {
  int n = blockIdx.x * blockDim.x + threadIdx.x;
  if (n < N_SAMP) atomicAdd(&meta[idxs[n]], 1);
}

__global__ void offsets_kernel(int* __restrict__ meta) {
  if (threadIdx.x == 0) {
    int off = 0;
    for (int k = 0; k < NK; ++k) {
      meta[16 + k] = off;
      off += (meta[k] + 15) & ~15;   // 16-align every bucket start
    }
    meta[24] = off;                  // padded total
  }
}

__global__ void scatter_kernel(const int* __restrict__ idxs, int* __restrict__ meta,
                               int* __restrict__ perm) {
  int n = blockIdx.x * blockDim.x + threadIdx.x;
  if (n < N_SAMP) {
    int k = idxs[n];
    int pos = meta[16 + k] + atomicAdd(&meta[8 + k], 1);
    perm[pos] = n;
  }
}

// ---------------------------------------------------------------------------
// mlp: one wave per 16-sample tile (single weight set k per tile).
//   L0: A = x tile (16x16 f32 -> f16, K padded to 32), 4 WMMA
//   L1: A from LDS-transposed h0, 2 K-steps x 4 N-tiles = 8 WMMA
//   L2: VALU dot(64) + shfl_xor reduce
// ---------------------------------------------------------------------------
__global__ __launch_bounds__(128) void mlp_kernel(
    const float* __restrict__ xs, const float* __restrict__ b0,
    const float* __restrict__ b1, const float* __restrict__ W2,
    const float* __restrict__ b2, const int* __restrict__ meta,
    const int* __restrict__ perm, const _Float16* __restrict__ W0p,
    const _Float16* __restrict__ W1p, float* __restrict__ out) {
  __shared__ __align__(16) _Float16 h0s[4][16][72];   // 72: pad to dodge bank conflicts
  const int lane = threadIdx.x & 31;
  const int wave = threadIdx.x >> 5;
  const int tile = blockIdx.x * 4 + wave;
  const int s = tile * 16;
  if (s >= meta[24]) return;                     // wave-uniform: past padded end
  int kk = 0;                                    // bucket (= weight set) of this tile
  while (kk < NK - 1 && s >= meta[16 + kk + 1]) ++kk;

  const int m15 = lane & 15;
  const int hi  = (lane & 16) ? 1 : 0;
  const int row = perm[s + m15];                 // -1 => dummy padding row

  // ---- layer 0 A fragment (A 16x32 f16: M=lane&15, e<8 -> K=8*hi+e, e>=8 -> K>=16 = pad)
  v16h a0 = {};
  if (row >= 0) {
    const float* xr = xs + (long)row * D_IN + (hi ? 8 : 0);
    #pragma unroll
    for (int e = 0; e < 8; ++e) a0[e] = (_Float16)xr[e];
  }

  const v16h* W0v = (const v16h*)W0p;
  const v16h* W1v = (const v16h*)W1p;

  // ---- layer 0: h0 = relu(x @ W0[k] + b0[k]) -> LDS (C layout -> [M][64])
  #pragma unroll
  for (int nt = 0; nt < 4; ++nt) {
    v8f acc = {};
    v16h b = W0v[(kk*4 + nt)*32 + lane];
    acc = wmma_f16(a0, b, acc);
    float bias = b0[kk*WIDTH + nt*16 + m15];
    #pragma unroll
    for (int r = 0; r < 8; ++r) {                // C: VGPR r -> M = r + 8*hi, N = lane&15
      float v = acc[r] + bias;
      v = v > 0.f ? v : 0.f;
      h0s[wave][hi*8 + r][nt*16 + m15] = (_Float16)v;
    }
  }
  asm volatile("s_wait_dscnt 0" ::: "memory");   // wave-local LDS RAW fence

  // ---- layer 1: h1 = relu(h0 @ W1[k] + b1[k])
  v8f zero = {};
  v8f c1[4];
  #pragma unroll
  for (int nt = 0; nt < 4; ++nt) c1[nt] = zero;
  #pragma unroll
  for (int kt = 0; kt < 2; ++kt) {
    const v8h* rowp = (const v8h*)&h0s[wave][m15][0];
    v8h r1 = rowp[(kt*32 + hi*8) >> 3];          // A elems 0..7:  K = kt*32 + 8*hi + e
    v8h r2 = rowp[(kt*32 + 16 + hi*8) >> 3];     // A elems 8..15: K = kt*32 + 16 + 8*hi + e
    v16h a1;
    #pragma unroll
    for (int e = 0; e < 8; ++e) { a1[e] = r1[e]; a1[e + 8] = r2[e]; }
    #pragma unroll
    for (int nt = 0; nt < 4; ++nt) {
      v16h b = W1v[((kk*2 + kt)*4 + nt)*32 + lane];
      c1[nt] = wmma_f16(a1, b, c1[nt]);
    }
  }

  // ---- layer 2: out = h1 @ W2[k] + b2[k]  (width 1 -> VALU + shfl reduce)
  float part[8];
  #pragma unroll
  for (int r = 0; r < 8; ++r) part[r] = 0.f;
  #pragma unroll
  for (int nt = 0; nt < 4; ++nt) {
    int ncol = nt*16 + m15;
    float w2 = W2[kk*WIDTH + ncol];
    float bb = b1[kk*WIDTH + ncol];
    #pragma unroll
    for (int r = 0; r < 8; ++r) {
      float h = c1[nt][r] + bb;
      h = h > 0.f ? h : 0.f;
      part[r] += h * w2;
    }
  }
  #pragma unroll
  for (int m = 1; m < 16; m <<= 1) {             // reduce over the 16 N-lanes per half
    #pragma unroll
    for (int r = 0; r < 8; ++r) part[r] += __shfl_xor(part[r], m, 32);
  }
  if (m15 == 0) {                                // lane 0 -> M 0..7, lane 16 -> M 8..15
    float bias2 = b2[kk];
    #pragma unroll
    for (int r = 0; r < 8; ++r) {
      int g = perm[s + hi*8 + r];
      if (g >= 0) out[g] = part[r] + bias2;
    }
  }
}

extern "C" void kernel_launch(void* const* d_in, const int* in_sizes, int n_in,
                              void* d_out, int out_size, void* d_ws, size_t ws_size,
                              hipStream_t stream) {
  const int*   idxs = (const int*)  d_in[0];
  const float* xs   = (const float*)d_in[1];
  const float* W0   = (const float*)d_in[2];
  const float* b0   = (const float*)d_in[3];
  const float* W1   = (const float*)d_in[4];
  const float* b1   = (const float*)d_in[5];
  const float* W2   = (const float*)d_in[6];
  const float* b2   = (const float*)d_in[7];
  float* out = (float*)d_out;

  char* ws = (char*)d_ws;
  int* meta = (int*)ws;                               // [0..7] counts, [8..15] cursors,
                                                      // [16..24] aligned offsets
  int* perm = (int*)(ws + 128);                       // NPAD ints
  _Float16* W0p = (_Float16*)(ws + 128 + sizeof(int)*(size_t)NPAD);  // 32 KB, 32B aligned
  _Float16* W1p = W0p + NK*4*32*16;                                  // 64 KB

  prep_kernel   <<<(NPAD + 255)/256, 256, 0, stream>>>(W0, W1, meta, perm, W0p, W1p);
  count_kernel  <<<N_SAMP/256,       256, 0, stream>>>(idxs, meta);
  offsets_kernel<<<1,                 32, 0, stream>>>(meta);
  scatter_kernel<<<N_SAMP/256,       256, 0, stream>>>(idxs, meta, perm);
  mlp_kernel    <<<MAXTILE/4,        128, 0, stream>>>(xs, b0, b1, W2, b2, meta, perm,
                                                       W0p, W1p, out);
}
